// seqLstm_25786983645437
// MI455X (gfx1250) — matrix operand order repaired and checked by reading, not compiled, not run on hardware
//
#include <hip/hip_runtime.h>

// 2-layer fused LSTM for MI455X (gfx1250, wave32, WMMA).
// B=4096, T=512, I=H=18. One wave owns 16 batch rows for the whole sequence.
//
// GEMM mapping per timestep (per wave):  gates^T[72x16] = W[72x19] * in^T[19x16]
//   M = gate index (5 tiles: i0-15, f0-15, g0-15, o0-15, {i,f,g,o}{16,17})
//   N = 16 batch rows,  K = 18 inputs + 1 bias column, zero padded to 32.
// D-layout => lane = batch, accumulator regs = gate rows: elementwise LSTM
// update is per-lane VALU. h is kept in B-operand layout; rebuilt each step
// with a single half-wave shfl_xor(16) exchange.

typedef __attribute__((ext_vector_type(16))) _Float16 v16h;
typedef __attribute__((ext_vector_type(8)))  float    v8f;

#define I_DIM 18
#define H_DIM 18
#define T_DIM 512
#define B_DIM 4096
#define NTILE 5
#define WAVES_PER_BLOCK 4
#define THREADS_PER_BLOCK (WAVES_PER_BLOCK * 32)
#define NUM_BLOCKS (B_DIM / (16 * WAVES_PER_BLOCK))

// CDNA5 has a hardware V_TANH_F32 trans op; use it if the toolchain exposes
// the builtin (guarded; falls back to the v_exp_f32 formulation otherwise).
#if __has_builtin(__builtin_amdgcn_tanhf)
  #define HW_TANH(x) __builtin_amdgcn_tanhf(x)
#elif __has_builtin(__builtin_amdgcn_tanh_f32)
  #define HW_TANH(x) __builtin_amdgcn_tanh_f32(x)
#endif

__device__ __forceinline__ float ftanh(float x) {
#ifdef HW_TANH
  return HW_TANH(x);                       // 1 trans op
#else
  return 1.0f - 2.0f / (__expf(2.0f * x) + 1.0f);
#endif
}
__device__ __forceinline__ float fsig(float x) {
#ifdef HW_TANH
  return __builtin_fmaf(0.5f, HW_TANH(0.5f * x), 0.5f);  // mul + tanh + fma
#else
  return 1.0f / (1.0f + __expf(-x));
#endif
}

// Weight element for A-operand: row G of W (72x18 row-major), column k.
// k==18 is the folded bias column (only for the W_ih operand).
__device__ __forceinline__ float wval(const float* __restrict__ W,
                                      const float* __restrict__ bih,
                                      const float* __restrict__ bhh,
                                      int G, int k) {
  if (G < 0) return 0.0f;
  if (k < I_DIM) return W[G * I_DIM + k];
  if (k == I_DIM && bih != nullptr) return bih[G] + bhh[G];
  return 0.0f;
}

// Build the 5 gate M-tiles of one weight matrix as WMMA A-operands.
// A 16x32 f16 layout used here: lane<16 holds row M=lane, K=0..15 (v0-7);
// lane>=16 holds row M=lane-16, K=16..31. Element 2v = even K (lo16).
__device__ __forceinline__ void build_A(const float* __restrict__ W,
                                        const float* __restrict__ bih,
                                        const float* __restrict__ bhh,
                                        int lane, v16h A[NTILE]) {
  const int half = lane >> 4;
  const int M = lane & 15;
  #pragma unroll
  for (int m = 0; m < NTILE; ++m) {
    int G;
    if (m < 4) G = m * H_DIM + M;                        // i,f,g,o rows 0..15
    else       G = (M < 8) ? (16 + (M & 1) + (M >> 1) * H_DIM) : -1; // 16,17 rows
    v16h a = {};
    #pragma unroll
    for (int v = 0; v < 8; ++v) {
      const int k0 = half * 16 + 2 * v;
      a[2 * v]     = (_Float16)wval(W, bih, bhh, G, k0);
      a[2 * v + 1] = (_Float16)wval(W, bih, bhh, G, k0 + 1);
    }
    A[m] = a;
  }
}

// One LSTM cell step for 16 batch rows. inB/hB are B-operands (K x 16).
// c[8] / hn[8] hold hidden units j = r + 8*half per lane; ce/h16/h17 hold
// j=16,17 (valid in lanes 0-15). Rebuilds hB (with 1.0 bias slot at K=18).
__device__ __forceinline__ void lstm_cell(const v16h* __restrict__ Aih,
                                          const v16h* __restrict__ Ahh,
                                          const v16h inB, v16h& hB,
                                          float c[8], float ce[2], int half,
                                          float hn[8], float& h16, float& h17) {
  v8f acc[NTILE];
  #pragma unroll
  for (int m = 0; m < NTILE; ++m) {
    v8f z = {};
    z = __builtin_amdgcn_wmma_f32_16x16x32_f16(false, Aih[m], false, inB,
                                               (short)0, z, false, false);
    z = __builtin_amdgcn_wmma_f32_16x16x32_f16(false, Ahh[m], false, hB,
                                               (short)0, z, false, false);
    acc[m] = z;
  }
  // Main 16 hidden units: per-lane, fully aligned across the 4 gate tiles.
  #pragma unroll
  for (int r = 0; r < 8; ++r) {
    const float ig = fsig(acc[0][r]);
    const float fg = fsig(acc[1][r]);
    const float gg = ftanh(acc[2][r]);
    const float og = fsig(acc[3][r]);
    const float cn = fg * c[r] + ig * gg;
    c[r] = cn;
    hn[r] = og * ftanh(cn);
  }
  // Tail tile: lanes 0-15 see {i16,i17,f16,f17,g16,g17,o16,o17} in acc[4].
  {
    const float i16 = fsig(acc[4][0]), i17 = fsig(acc[4][1]);
    const float f16 = fsig(acc[4][2]), f17 = fsig(acc[4][3]);
    const float g16 = ftanh(acc[4][4]), g17 = ftanh(acc[4][5]);
    const float o16 = fsig(acc[4][6]), o17 = fsig(acc[4][7]);
    ce[0] = f16 * ce[0] + i16 * g16;
    ce[1] = f17 * ce[1] + i17 * g17;
    h16 = o16 * ftanh(ce[0]);
    h17 = o17 * ftanh(ce[1]);
  }
  // Rebuild h as a B-operand: lane<16 needs K=j=0..15 (j=8..15 live in the
  // partner lane), lane>=16 needs K=16..18 -> (h16,h17,1.0).
  float p[8];
  #pragma unroll
  for (int r = 0; r < 8; ++r) p[r] = __shfl_xor(hn[r], 16, 32);
  const float q16 = __shfl_xor(h16, 16, 32);
  const float q17 = __shfl_xor(h17, 16, 32);
  v16h nb = {};
  if (half == 0) {
    #pragma unroll
    for (int v = 0; v < 4; ++v) {
      nb[2 * v]     = (_Float16)hn[2 * v];
      nb[2 * v + 1] = (_Float16)hn[2 * v + 1];
    }
    #pragma unroll
    for (int v = 4; v < 8; ++v) {
      nb[2 * v]     = (_Float16)p[2 * (v - 4)];
      nb[2 * v + 1] = (_Float16)p[2 * (v - 4) + 1];
    }
  } else {
    nb[0] = (_Float16)q16;
    nb[1] = (_Float16)q17;
    nb[2] = (_Float16)1.0f;  // K=18 bias column for the next layer's W_ih
  }
  hB = nb;
}

__global__ __launch_bounds__(THREADS_PER_BLOCK, 1)
void seqLstm_fused_kernel(const float* __restrict__ x,
                          const float* __restrict__ wih0, const float* __restrict__ whh0,
                          const float* __restrict__ bih0, const float* __restrict__ bhh0,
                          const float* __restrict__ wih1, const float* __restrict__ whh1,
                          const float* __restrict__ bih1, const float* __restrict__ bhh1,
                          float* __restrict__ out) {
  const int lane = threadIdx.x & 31;
  const int wave = threadIdx.x >> 5;
  const int half = lane >> 4;
  const int n    = lane & 15;
  const int b0   = (blockIdx.x * WAVES_PER_BLOCK + wave) * 16;

  // Pre-swizzled weight A-operands, resident in registers for the whole run.
  v16h Aih0[NTILE], Ahh0[NTILE], Aih1[NTILE], Ahh1[NTILE];
  build_A(wih0, bih0, bhh0, lane, Aih0);
  build_A(whh0, nullptr, nullptr, lane, Ahh0);   // K=18 column stays zero
  build_A(wih1, bih1, bhh1, lane, Aih1);
  build_A(whh1, nullptr, nullptr, lane, Ahh1);

  float c0[8], c1[8], c0e[2], c1e[2];
  #pragma unroll
  for (int r = 0; r < 8; ++r) { c0[r] = 0.0f; c1[r] = 0.0f; }
  c0e[0] = c0e[1] = c1e[0] = c1e[1] = 0.0f;

  v16h hB0 = {}, hB1 = {};
  if (half) { hB0[2] = (_Float16)1.0f; hB1[2] = (_Float16)1.0f; }  // bias slot

  const float* __restrict__ xr   = x   + (size_t)(b0 + n) * T_DIM * I_DIM;
  float* __restrict__       outr = out + (size_t)(b0 + n) * T_DIM * H_DIM;

  #pragma unroll 1
  for (int t = 0; t < T_DIM; ++t) {
    const float* xp = xr + t * I_DIM;
    // x^T B-operand: lane<16 -> K=0..15 of its batch row; lane>=16 -> K=16,17 + bias 1.0
    v16h xB = {};
    if (half == 0) {
      #pragma unroll
      for (int v = 0; v < 8; ++v) {
        const float2 w = *(const float2*)(xp + 2 * v);
        xB[2 * v]     = (_Float16)w.x;
        xB[2 * v + 1] = (_Float16)w.y;
      }
    } else {
      const float2 w = *(const float2*)(xp + 16);
      xB[0] = (_Float16)w.x;
      xB[1] = (_Float16)w.y;
      xB[2] = (_Float16)1.0f;
    }
    if (t + 16 < T_DIM) __builtin_prefetch(xp + 16 * I_DIM, 0, 3);

    float h0n[8], h0a, h0b;
    lstm_cell(Aih0, Ahh0, xB, hB0, c0, c0e, half, h0n, h0a, h0b);
    float h1n[8], h1a, h1b;
    lstm_cell(Aih1, Ahh1, hB0, hB1, c1, c1e, half, h1n, h1a, h1b);  // layer-1 input = fresh h0

    float* op = outr + t * H_DIM;
    if (half == 0) {
      #pragma unroll
      for (int v = 0; v < 4; ++v) {
        float2 s; s.x = h1n[2 * v]; s.y = h1n[2 * v + 1];
        *(float2*)(op + 2 * v * 2) = s;          // j = 0..7
      }
      float2 s; s.x = h1a; s.y = h1b;
      *(float2*)(op + 16) = s;                   // j = 16,17
    } else {
      #pragma unroll
      for (int v = 0; v < 4; ++v) {
        float2 s; s.x = h1n[2 * v]; s.y = h1n[2 * v + 1];
        *(float2*)(op + 8 + 2 * v * 2) = s;      // j = 8..15
      }
    }
  }
}

extern "C" void kernel_launch(void* const* d_in, const int* in_sizes, int n_in,
                              void* d_out, int out_size, void* d_ws, size_t ws_size,
                              hipStream_t stream) {
  (void)in_sizes; (void)n_in; (void)d_ws; (void)ws_size; (void)out_size;
  const float* x    = (const float*)d_in[0];
  const float* wih0 = (const float*)d_in[1];
  const float* whh0 = (const float*)d_in[2];
  const float* bih0 = (const float*)d_in[3];
  const float* bhh0 = (const float*)d_in[4];
  const float* wih1 = (const float*)d_in[5];
  const float* whh1 = (const float*)d_in[6];
  const float* bih1 = (const float*)d_in[7];
  const float* bhh1 = (const float*)d_in[8];
  float* out = (float*)d_out;

  seqLstm_fused_kernel<<<NUM_BLOCKS, THREADS_PER_BLOCK, 0, stream>>>(
      x, wih0, whh0, bih0, bhh0, wih1, whh1, bih1, bhh1, out);
}
